// LlamaAttention_7267084665266
// MI455X (gfx1250) — matrix operand order repaired and checked by reading, not compiled, not run on hardware
//
#include <hip/hip_runtime.h>

// ---------------------------------------------------------------------------
// MLA attention for gfx1250 (MI455X): bf16 WMMA everywhere, fp32 accumulate,
// flash-style softmax (no SxS materialization). Compute-bound problem ->
// keep everything on v_wmma_f32_16x16x32_bf16, with multi-accumulator GEMM
// tiles (32x64 per wave) so the matrix pipe isn't serialized on one
// accumulator chain.
// ---------------------------------------------------------------------------

typedef __bf16 bf16_t;
typedef __attribute__((ext_vector_type(16))) __bf16 v16bf;
typedef __attribute__((ext_vector_type(8)))  float  v8f;
typedef __attribute__((ext_vector_type(4)))  unsigned int v4u;

#define DIMN    2048
#define HEADS   16
#define HALF    64
#define HEAD_D  128
#define LATENT  256
#define BATCH   2
#define SEQ     2048
#define ROWS    (BATCH * SEQ)   // 4096

union FragU { v16bf v; v4u u[2]; };

// A-fragment (16x32, MxK): lane m = lane%16; half h = lane/16 picks
// K chunks [8h .. 8h+7] and [16+8h .. 16+8h+7]. Row must be K-contiguous.
__device__ __forceinline__ v16bf load_fragA(const bf16_t* row, int k0, int lane) {
    int base = k0 + ((lane >> 4) << 3);
    FragU f;
    f.u[0] = *(const v4u*)(row + base);
    f.u[1] = *(const v4u*)(row + base + 16);
    return f.v;
}

// B-fragment (32x16, KxN): lane n = lane%16; half h = lane/16 picks
// K chunks [8h .. 8h+7] and [16+8h .. 16+8h+7]. "row" = column n of B, K-contiguous
// (i.e. row n of B^T).
__device__ __forceinline__ v16bf load_fragB(const bf16_t* row, int k0, int lane) {
    int base = k0 + ((lane >> 4) << 3);
    FragU f;
    f.u[0] = *(const v4u*)(row + base);
    f.u[1] = *(const v4u*)(row + base + 16);
    return f.v;
}

__device__ __forceinline__ v8f wmma_bf16(v16bf a, v16bf b, v8f c) {
    return __builtin_amdgcn_wmma_f32_16x16x32_bf16(
        false, a, false, b, (short)0, c, false, false);
}

// ---------------------------------------------------------------------------
// Elementwise prep kernels
// ---------------------------------------------------------------------------
__global__ void cast_bf16_kernel(const float* __restrict__ src,
                                 bf16_t* __restrict__ dst, int n) {
    int i = blockIdx.x * blockDim.x + threadIdx.x;
    if (i < n) dst[i] = (bf16_t)src[i];
}

// src [K][N] fp32 -> dst [N][K] bf16
__global__ void transpose_cast_kernel(const float* __restrict__ src,
                                      bf16_t* __restrict__ dst, int K, int N) {
    int i = blockIdx.x * blockDim.x + threadIdx.x;
    if (i >= K * N) return;
    int k = i / N, n = i - k * N;
    dst[(size_t)n * K + k] = (bf16_t)src[i];
}

// In-place RoPE on t[ROWS][HEADS*HALF] bf16. One thread per (row, head, freq).
__global__ void rope_kernel(bf16_t* __restrict__ t) {
    int i = blockIdx.x * blockDim.x + threadIdx.x;   // ROWS*HEADS*32
    if (i >= ROWS * HEADS * 32) return;
    int fi  = i & 31;
    int h   = (i >> 5) & (HEADS - 1);
    int row = i >> 9;
    int s   = row & (SEQ - 1);
    float inv_freq = powf(10000.0f, -(float)fi * (1.0f / 32.0f));
    float sn, cs;
    sincosf((float)s * inv_freq, &sn, &cs);
    bf16_t* p = t + (size_t)row * (HEADS * HALF) + h * HALF + fi;
    float a = (float)p[0];
    float b = (float)p[32];
    p[0]  = (bf16_t)(a * cs - b * sn);
    p[32] = (bf16_t)(b * cs + a * sn);
}

// ---------------------------------------------------------------------------
// Generic bf16 GEMM: C[M,N] = A[M,K] @ B (B given transposed: Bt[N][K]).
// One wave computes a 32x64 strip: 2 A-fragments x 4 B-fragments ->
// 8 independent accumulators (breaks the WMMA dependency chain) and
// 1.5 b128 loads per WMMA. mode 0: fp32 C; mode 1: bf16 C;
// mode 2: bf16 scatter into vT[B][H][128][S] layout.
// Requires M % 32 == 0, N % 64 == 0, K % 32 == 0.
// ---------------------------------------------------------------------------
__global__ __launch_bounds__(256)
void gemm_bf16_kernel(const bf16_t* __restrict__ A, const bf16_t* __restrict__ Bt,
                      void* __restrict__ Cout, int M, int N, int K, int mode) {
    int lane = threadIdx.x & 31;
    int wave = threadIdx.x >> 5;
    int tile = blockIdx.x * 8 + wave;
    int ntn  = N >> 6;                       // strips of 64 in N
    int total = (M >> 5) * ntn;
    if (tile >= total) return;               // wave-uniform: EXEC stays full
    int mt = tile / ntn;
    int nt = tile - mt * ntn;
    int ln = lane & 15;

    const bf16_t* arow0 = A + (size_t)(mt * 32 + ln) * K;
    const bf16_t* arow1 = arow0 + (size_t)16 * K;
    const bf16_t* brow[4];
    #pragma unroll
    for (int t = 0; t < 4; ++t)
        brow[t] = Bt + (size_t)(nt * 64 + t * 16 + ln) * K;

    v8f acc[2][4];
    #pragma unroll
    for (int s = 0; s < 2; ++s)
        #pragma unroll
        for (int t = 0; t < 4; ++t)
            acc[s][t] = (v8f){0.f,0.f,0.f,0.f,0.f,0.f,0.f,0.f};

    for (int k0 = 0; k0 < K; k0 += 32) {
        v16bf a0 = load_fragA(arow0, k0, lane);
        v16bf a1 = load_fragA(arow1, k0, lane);
        #pragma unroll
        for (int t = 0; t < 4; ++t) {
            v16bf b = load_fragB(brow[t], k0, lane);
            acc[0][t] = wmma_bf16(a0, b, acc[0][t]);
            acc[1][t] = wmma_bf16(a1, b, acc[1][t]);
        }
    }

    int half = lane >> 4;
    if (mode == 0) {
        float* C = (float*)Cout;
        #pragma unroll
        for (int s = 0; s < 2; ++s)
            #pragma unroll
            for (int t = 0; t < 4; ++t) {
                int n = nt * 64 + t * 16 + ln;
                #pragma unroll
                for (int r = 0; r < 8; ++r) {
                    int m = mt * 32 + s * 16 + r + 8 * half;
                    C[(size_t)m * N + n] = acc[s][t][r];
                }
            }
    } else if (mode == 1) {
        bf16_t* C = (bf16_t*)Cout;
        #pragma unroll
        for (int s = 0; s < 2; ++s)
            #pragma unroll
            for (int t = 0; t < 4; ++t) {
                int n = nt * 64 + t * 16 + ln;
                #pragma unroll
                for (int r = 0; r < 8; ++r) {
                    int m = mt * 32 + s * 16 + r + 8 * half;
                    C[(size_t)m * N + n] = (bf16_t)acc[s][t][r];
                }
            }
    } else {
        // v scatter: row m = b*SEQ+s, col n = h*128+dd  ->  vT[b][h][dd][s]
        bf16_t* C = (bf16_t*)Cout;
        #pragma unroll
        for (int s = 0; s < 2; ++s)
            #pragma unroll
            for (int t = 0; t < 4; ++t) {
                int n  = nt * 64 + t * 16 + ln;
                int h  = n >> 7;
                int dd = n & 127;
                #pragma unroll
                for (int r = 0; r < 8; ++r) {
                    int m  = mt * 32 + s * 16 + r + 8 * half;
                    int bb = m >> 11;
                    int sq = m & (SEQ - 1);
                    C[((size_t)((bb * HEADS + h) * HEAD_D + dd) << 11) + sq] =
                        (bf16_t)acc[s][t][r];
                }
            }
    }
}

// ---------------------------------------------------------------------------
// Flash attention. q,k: bf16 [ROWS][HEADS*64] (RoPE already applied);
// vT: bf16 [B][H][128][S]; o: bf16 [ROWS][DIM].
// Block = 256 threads = 8 waves; each wave owns 16 q rows of one (b,h).
// ---------------------------------------------------------------------------
__global__ __launch_bounds__(256)
void attn_kernel(const bf16_t* __restrict__ q, const bf16_t* __restrict__ k,
                 const bf16_t* __restrict__ vT, bf16_t* __restrict__ o) {
    __shared__ alignas(16) bf16_t plds[8][16][32];   // per-wave P tile

    int lane = threadIdx.x & 31;
    int wave = threadIdx.x >> 5;
    int half = lane >> 4;
    int ln   = lane & 15;

    int bh   = blockIdx.x >> 4;        // 0..31
    int qblk = blockIdx.x & 15;
    int b    = bh >> 4;                // 0..1
    int h    = bh & 15;
    int q0   = qblk * 128 + wave * 16;

    const bf16_t* qrow = q + (size_t)(b * SEQ + q0 + ln) * (HEADS * HALF) + h * HALF;
    v16bf qf0 = load_fragA(qrow, 0, lane);
    v16bf qf1 = load_fragA(qrow, 32, lane);

    const bf16_t* kbase = k  + (size_t)(b * SEQ) * (HEADS * HALF) + h * HALF;
    const bf16_t* vbase = vT + (size_t)(b * HEADS + h) * HEAD_D * SEQ;

    float m_run[8], l_run[8];
    v8f oacc[8];
    #pragma unroll
    for (int r = 0; r < 8; ++r) { m_run[r] = -1e30f; l_run[r] = 0.f; }
    #pragma unroll
    for (int t = 0; t < 8; ++t) oacc[t] = (v8f){0.f,0.f,0.f,0.f,0.f,0.f,0.f,0.f};

    for (int j = 0; j < SEQ; j += 32) {
        // ---- logits for 16 q rows x 32 kv columns (two 16-wide N tiles) ----
        float sv[2][8];
        #pragma unroll
        for (int t = 0; t < 2; ++t) {
            const bf16_t* krow = kbase + (size_t)(j + t * 16 + ln) * (HEADS * HALF);
            v16bf kf0 = load_fragB(krow, 0, lane);
            v16bf kf1 = load_fragB(krow, 32, lane);
            v8f s = {0.f,0.f,0.f,0.f,0.f,0.f,0.f,0.f};
            s = wmma_bf16(qf0, kf0, s);
            s = wmma_bf16(qf1, kf1, s);
            #pragma unroll
            for (int r = 0; r < 8; ++r) sv[t][r] = s[r] * 0.125f;  // 1/sqrt(64)
        }

        // ---- online softmax; row m = r + 8*half lives across the 16 lanes
        //      of this half, so reduce with xor-shuffles <16 ----
        float alpha[8];
        #pragma unroll
        for (int r = 0; r < 8; ++r) {
            float tmax = fmaxf(sv[0][r], sv[1][r]);
            #pragma unroll
            for (int off = 1; off < 16; off <<= 1)
                tmax = fmaxf(tmax, __shfl_xor(tmax, off, 32));
            float nm = fmaxf(m_run[r], tmax);
            float p0 = __expf(sv[0][r] - nm);
            float p1 = __expf(sv[1][r] - nm);
            float tsum = p0 + p1;
            #pragma unroll
            for (int off = 1; off < 16; off <<= 1)
                tsum += __shfl_xor(tsum, off, 32);
            alpha[r] = __expf(m_run[r] - nm);
            l_run[r] = l_run[r] * alpha[r] + tsum;
            m_run[r] = nm;
            sv[0][r] = p0; sv[1][r] = p1;
        }

        // ---- re-layout P (D-layout) -> A-fragment via per-wave LDS slice ----
        #pragma unroll
        for (int t = 0; t < 2; ++t)
            #pragma unroll
            for (int r = 0; r < 8; ++r)
                plds[wave][r + 8 * half][t * 16 + ln] = (bf16_t)sv[t][r];
        __syncthreads();   // uniform; also forces DS waits
        v16bf pf = load_fragA(&plds[wave][ln][0], 0, lane);
        __syncthreads();

        // ---- rescale O and accumulate P @ V over the 128-dim head ----
        #pragma unroll
        for (int t = 0; t < 8; ++t) {
            #pragma unroll
            for (int r = 0; r < 8; ++r) oacc[t][r] = oacc[t][r] * alpha[r];
            const bf16_t* vrow = vbase + (size_t)(t * 16 + ln) * SEQ + j;
            v16bf vf = load_fragB(vrow, 0, lane);
            oacc[t] = wmma_bf16(pf, vf, oacc[t]);
        }
    }

    // ---- epilogue: normalize and store bf16 [ROWS][DIM] ----
    #pragma unroll
    for (int t = 0; t < 8; ++t) {
        #pragma unroll
        for (int r = 0; r < 8; ++r) {
            int m   = q0 + r + 8 * half;
            int col = h * HEAD_D + t * 16 + ln;
            o[(size_t)(b * SEQ + m) * DIMN + col] =
                (bf16_t)(oacc[t][r] / l_run[r]);
        }
    }
}

// ---------------------------------------------------------------------------
// Host launcher
// ---------------------------------------------------------------------------
extern "C" void kernel_launch(void* const* d_in, const int* in_sizes, int n_in,
                              void* d_out, int out_size, void* d_ws, size_t ws_size,
                              hipStream_t stream) {
    const float* x     = (const float*)d_in[0];
    const float* wq_d  = (const float*)d_in[1];
    const float* wkv_d = (const float*)d_in[2];
    const float* wq_u  = (const float*)d_in[3];
    const float* wk_u  = (const float*)d_in[4];
    const float* wv_u  = (const float*)d_in[5];
    const float* wo    = (const float*)d_in[6];
    float* out = (float*)d_out;

    char* ws = (char*)d_ws;
    auto alloc = [&](size_t elems) {
        bf16_t* p = (bf16_t*)ws;
        ws += elems * sizeof(bf16_t);
        return p;
    };
    bf16_t* x_bf   = alloc((size_t)ROWS * DIMN);          // 16.8 MB
    bf16_t* wqd_t  = alloc((size_t)LATENT * DIMN);        // Bt [256][2048]
    bf16_t* wkvd_t = alloc((size_t)LATENT * DIMN);
    bf16_t* wqu_t  = alloc((size_t)HEADS * HALF * LATENT);// Bt [1024][256]
    bf16_t* wku_t  = alloc((size_t)HEADS * HALF * LATENT);
    bf16_t* wvu_t  = alloc((size_t)DIMN * LATENT);        // Bt [2048][256]
    bf16_t* wo_t   = alloc((size_t)DIMN * DIMN);          // Bt [2048][2048]
    bf16_t* qlat   = alloc((size_t)ROWS * LATENT);
    bf16_t* kvlat  = alloc((size_t)ROWS * LATENT);
    bf16_t* qraw   = alloc((size_t)ROWS * HEADS * HALF);
    bf16_t* kraw   = alloc((size_t)ROWS * HEADS * HALF);
    bf16_t* vT     = alloc((size_t)BATCH * HEADS * HEAD_D * SEQ);
    bf16_t* attnO  = alloc((size_t)ROWS * DIMN);
    // total ~84 MB of workspace

    const int TB = 256;
    auto blocks = [](int n) { return (n + 255) / 256; };
    // 32x64 tiles per wave, 8 waves per block
    auto gemm_blocks = [](int M, int N) { return ((M / 32) * (N / 64) + 7) / 8; };

    // prep: casts + weight transposes
    cast_bf16_kernel<<<blocks(ROWS * DIMN), TB, 0, stream>>>(x, x_bf, ROWS * DIMN);
    transpose_cast_kernel<<<blocks(DIMN * LATENT), TB, 0, stream>>>(wq_d,  wqd_t,  DIMN, LATENT);
    transpose_cast_kernel<<<blocks(DIMN * LATENT), TB, 0, stream>>>(wkv_d, wkvd_t, DIMN, LATENT);
    transpose_cast_kernel<<<blocks(LATENT * HEADS * HALF), TB, 0, stream>>>(wq_u, wqu_t, LATENT, HEADS * HALF);
    transpose_cast_kernel<<<blocks(LATENT * HEADS * HALF), TB, 0, stream>>>(wk_u, wku_t, LATENT, HEADS * HALF);
    transpose_cast_kernel<<<blocks(LATENT * DIMN), TB, 0, stream>>>(wv_u, wvu_t, LATENT, DIMN);
    transpose_cast_kernel<<<blocks(DIMN * DIMN), TB, 0, stream>>>(wo, wo_t, DIMN, DIMN);

    // down-projections: [4096,2048] @ [2048,256]
    gemm_bf16_kernel<<<gemm_blocks(ROWS, LATENT), TB, 0, stream>>>(x_bf, wqd_t,  qlat,  ROWS, LATENT, DIMN, 1);
    gemm_bf16_kernel<<<gemm_blocks(ROWS, LATENT), TB, 0, stream>>>(x_bf, wkvd_t, kvlat, ROWS, LATENT, DIMN, 1);

    // up-projections: q/k [4096,256]@[256,1024], v [4096,256]@[256,2048] (vT scatter)
    gemm_bf16_kernel<<<gemm_blocks(ROWS, HEADS * HALF), TB, 0, stream>>>(qlat,  wqu_t, qraw, ROWS, HEADS * HALF, LATENT, 1);
    gemm_bf16_kernel<<<gemm_blocks(ROWS, HEADS * HALF), TB, 0, stream>>>(kvlat, wku_t, kraw, ROWS, HEADS * HALF, LATENT, 1);
    gemm_bf16_kernel<<<gemm_blocks(ROWS, DIMN), TB, 0, stream>>>(kvlat, wvu_t, vT, ROWS, DIMN, LATENT, 2);

    // RoPE in place on q and k
    rope_kernel<<<blocks(ROWS * HEADS * 32), TB, 0, stream>>>(qraw);
    rope_kernel<<<blocks(ROWS * HEADS * 32), TB, 0, stream>>>(kraw);

    // flash attention: 2 batches * 16 heads * 16 q-blocks of 128 rows
    attn_kernel<<<BATCH * HEADS * (SEQ / 128), TB, 0, stream>>>(qraw, kraw, vT, attnO);

    // output projection: [4096,2048] @ [2048,2048] -> fp32 d_out
    gemm_bf16_kernel<<<gemm_blocks(ROWS, DIMN), TB, 0, stream>>>(attnO, wo_t, out, ROWS, DIMN, DIMN, 0);
}
